// Attend_45440753992364
// MI455X (gfx1250) — matrix operand order repaired and checked by reading, not compiled, not run on hardware
//
#include <hip/hip_runtime.h>
#include <stdint.h>

typedef __attribute__((ext_vector_type(16))) __bf16 v16bf;
typedef __attribute__((ext_vector_type(8)))  __bf16 v8bf;
typedef __attribute__((ext_vector_type(8)))  float  v8f;

#define BHN 16
#define SEQ 4096
#define HD  128
#define NW  16    // waves per block
#define BQ  256   // query rows per block (16 waves x 16 rows) - halves K/V L2 re-reads
#define BK  64    // kv rows per tile iteration

// native f32 -> bf16 (hardware cvt on gfx1250)
static __device__ __forceinline__ __bf16 f2bf(float f) { return (__bf16)f; }

static __device__ __forceinline__ float fexp2(float x) {
#if __has_builtin(__builtin_amdgcn_exp2f)
  return __builtin_amdgcn_exp2f(x);   // v_exp_f32
#else
  return exp2f(x);
#endif
}

// max-reduction across the 16-lane half-groups (wave32)
static __device__ __forceinline__ float rowmax16(float x) {
  x = fmaxf(x, __shfl_xor(x, 1, 32));
  x = fmaxf(x, __shfl_xor(x, 2, 32));
  x = fmaxf(x, __shfl_xor(x, 4, 32));
  x = fmaxf(x, __shfl_xor(x, 8, 32));
  return x;
}

__global__ __launch_bounds__(512) void flash_attn_wmma_bf16(
    const float* __restrict__ q, const float* __restrict__ k,
    const float* __restrict__ v, float* __restrict__ out)
{
  // double-buffered K/V tiles + per-wave P relayout buffers (96 KB total)
  __shared__ __bf16 sK [2][BK][HD];    // K tile, row-major [kv][d]   (2 x 16 KB)
  __shared__ __bf16 sVt[2][HD][BK];    // V tile, transposed [d][kv]  (2 x 16 KB)
  __shared__ __bf16 sP [NW][16][BK];   // per-wave P buffers          (32 KB)

  const int tid  = threadIdx.x;
  const int wave = tid >> 5;
  const int lane = tid & 31;
  const int hsel = lane >> 4;   // half-wave select
  const int lr   = lane & 15;

  const int qtile = blockIdx.x;
  const int bh    = blockIdx.y;
  const int qbase = qtile * BQ + wave * 16;

  // 1/sqrt(128) * log2(e): base-2 softmax, scale folded into Q
  const float qscale = 0.08838834764831845f * 1.4426950408889634f;

  // ---- Q fragments: 16x128 bf16 in WMMA A-layout, scale folded in ----
  v16bf aq[4];
  const float* qrow = q + ((size_t)bh * SEQ + qbase + lr) * HD;
  #pragma unroll
  for (int c = 0; c < 4; ++c) {
    const int b1 = c * 32 + 16 * hsel;
    const float4 x0 = *(const float4*)(qrow + b1);
    const float4 x1 = *(const float4*)(qrow + b1 + 4);
    const float4 y0 = *(const float4*)(qrow + b1 + 8);
    const float4 y1 = *(const float4*)(qrow + b1 + 12);
    v16bf a;
    a[0]=f2bf(x0.x*qscale); a[1]=f2bf(x0.y*qscale); a[2]=f2bf(x0.z*qscale); a[3]=f2bf(x0.w*qscale);
    a[4]=f2bf(x1.x*qscale); a[5]=f2bf(x1.y*qscale); a[6]=f2bf(x1.z*qscale); a[7]=f2bf(x1.w*qscale);
    a[8]=f2bf(y0.x*qscale); a[9]=f2bf(y0.y*qscale); a[10]=f2bf(y0.z*qscale); a[11]=f2bf(y0.w*qscale);
    a[12]=f2bf(y1.x*qscale); a[13]=f2bf(y1.y*qscale); a[14]=f2bf(y1.z*qscale); a[15]=f2bf(y1.w*qscale);
    aq[c] = a;
  }

  // all-ones B fragment: O_l = alpha*O_l + P x 1 accumulates P row sums
  v16bf bones;
  #pragma unroll
  for (int i = 0; i < 16; ++i) bones[i] = (__bf16)1.0f;

  // ---- accumulators + online-softmax state (rows = r + 8*hsel) ----
  v8f o[8];
  #pragma unroll
  for (int i = 0; i < 8; ++i) o[i] = {};
  v8f ol = {};                 // row-sum accumulator (softmax denominator)
  float m[8];
  #pragma unroll
  for (int r = 0; r < 8; ++r) m[r] = -1.0e30f;

  // staging decomposition: 512 threads -> one (row 0..63, 16-col strip) each
  const int srow = tid >> 3;          // 0..63
  const int scol = (tid & 7) << 4;    // 0,16,...,112
  const float* kbase = k + ((size_t)bh * SEQ + srow) * HD + scol;
  const float* vbase = v + ((size_t)bh * SEQ + srow) * HD + scol;
  const int NT = SEQ / BK;            // 64 tiles

  // stage one 64x128 K/V tile (f32 global -> bf16 LDS, V transposed)
  auto stage = [&](int kt, int buf) {
    const float* kp = kbase + (size_t)kt * BK * HD;
    const float* vp = vbase + (size_t)kt * BK * HD;
    v8bf kb0, kb1;
    #pragma unroll
    for (int j = 0; j < 4; ++j) {
      const float4 kf = *(const float4*)(kp + j * 4);
      const float4 vf = *(const float4*)(vp + j * 4);
      v8bf& kb = (j < 2) ? kb0 : kb1;
      const int e = (j & 1) * 4;
      kb[e+0]=f2bf(kf.x); kb[e+1]=f2bf(kf.y); kb[e+2]=f2bf(kf.z); kb[e+3]=f2bf(kf.w);
      sVt[buf][scol + j*4 + 0][srow] = f2bf(vf.x);
      sVt[buf][scol + j*4 + 1][srow] = f2bf(vf.y);
      sVt[buf][scol + j*4 + 2][srow] = f2bf(vf.z);
      sVt[buf][scol + j*4 + 3][srow] = f2bf(vf.w);
    }
    *(v8bf*)(&sK[buf][srow][scol])     = kb0;   // 2 x ds_store_b128
    *(v8bf*)(&sK[buf][srow][scol + 8]) = kb1;
    if (kt + 1 < NT) {  // hint tile kt+1 toward caches (global_prefetch_b8)
      __builtin_prefetch(kp + BK * HD, 0, 0);
      __builtin_prefetch(vp + BK * HD, 0, 0);
    }
  };

  stage(0, 0);  // prologue: fill buffer 0

  for (int kt = 0; kt < NT; ++kt) {
    const int cur = kt & 1;
    __syncthreads();                          // tile kt staged; prev compute done
    if (kt + 1 < NT) stage(kt + 1, cur ^ 1);  // overlap next tile with compute

    // ---- GEMM1: S(16x64) = Q (16x128) x K^T, four 16x16 C tiles ----
    v8f s[4];
    #pragma unroll
    for (int t = 0; t < 4; ++t) s[t] = {};
    #pragma unroll
    for (int c = 0; c < 4; ++c) {
      #pragma unroll
      for (int t = 0; t < 4; ++t) {
        const v16bf b = *(const v16bf*)(&sK[cur][16*t + lr][c * 32 + 16 * hsel]);
        s[t] = __builtin_amdgcn_wmma_f32_16x16x32_bf16(false, aq[c], false, b,
                                                       (short)0, s[t], false, false);
      }
    }

    // ---- online softmax (base-2) over the 64 new columns ----
    float a8[8], p[4][8];
    #pragma unroll
    for (int r = 0; r < 8; ++r) {
      float mx = fmaxf(fmaxf(s[0][r], s[1][r]), fmaxf(s[2][r], s[3][r]));
      mx = rowmax16(mx);
      const float mi = fmaxf(m[r], mx);
      a8[r] = fexp2(m[r] - mi);
      m[r]  = mi;
      #pragma unroll
      for (int t = 0; t < 4; ++t) p[t][r] = fexp2(s[t][r] - mi);
    }

    // ---- relayout P: C-layout -> A-layout via private LDS (same-wave DS in-order) ----
    #pragma unroll
    for (int r = 0; r < 8; ++r) {
      #pragma unroll
      for (int t = 0; t < 4; ++t)
        sP[wave][r + 8 * hsel][16*t + lr] = f2bf(p[t][r]);
    }
    const v8bf pl0 = *(const v8bf*)(&sP[wave][lr][16 * hsel]);
    const v8bf ph0 = *(const v8bf*)(&sP[wave][lr][8 + 16 * hsel]);
    const v8bf pl1 = *(const v8bf*)(&sP[wave][lr][32 + 16 * hsel]);
    const v8bf ph1 = *(const v8bf*)(&sP[wave][lr][40 + 16 * hsel]);
    const v16bf ap0 = __builtin_shufflevector(pl0, ph0, 0,1,2,3,4,5,6,7,
                                              8,9,10,11,12,13,14,15);
    const v16bf ap1 = __builtin_shufflevector(pl1, ph1, 0,1,2,3,4,5,6,7,
                                              8,9,10,11,12,13,14,15);

    // ---- row sums on the matrix pipe: O_l = alpha*O_l + P x 1 ----
    {
      v8f oo = ol;
      #pragma unroll
      for (int r = 0; r < 8; ++r) oo[r] *= a8[r];
      oo = __builtin_amdgcn_wmma_f32_16x16x32_bf16(false, ap0, false, bones,
                                                   (short)0, oo, false, false);
      ol = __builtin_amdgcn_wmma_f32_16x16x32_bf16(false, ap1, false, bones,
                                                   (short)0, oo, false, false);
    }

    // ---- GEMM2: O = alpha*O + P(16x64) x V(64x128), 8 d-chunks x 2 K-chunks ----
    #pragma unroll
    for (int dc = 0; dc < 8; ++dc) {
      v8f oo = o[dc];
      #pragma unroll
      for (int r = 0; r < 8; ++r) oo[r] *= a8[r];
      const v16bf bv0 = *(const v16bf*)(&sVt[cur][dc * 16 + lr][16 * hsel]);
      const v16bf bv1 = *(const v16bf*)(&sVt[cur][dc * 16 + lr][32 + 16 * hsel]);
      oo = __builtin_amdgcn_wmma_f32_16x16x32_bf16(false, ap0, false, bv0,
                                                   (short)0, oo, false, false);
      o[dc] = __builtin_amdgcn_wmma_f32_16x16x32_bf16(false, ap1, false, bv1,
                                                      (short)0, oo, false, false);
    }
  }

  // ---- epilogue: normalize (8 rcps) and store coalesced ----
  float rinv[8];
  #pragma unroll
  for (int r = 0; r < 8; ++r) rinv[r] = 1.0f / ol[r];
  float* orow = out + ((size_t)bh * SEQ + qbase) * HD;
  #pragma unroll
  for (int dc = 0; dc < 8; ++dc) {
    #pragma unroll
    for (int r = 0; r < 8; ++r) {
      orow[(size_t)(r + 8 * hsel) * HD + dc * 16 + lr] = o[dc][r] * rinv[r];
    }
  }
}

extern "C" void kernel_launch(void* const* d_in, const int* in_sizes, int n_in,
                              void* d_out, int out_size, void* d_ws, size_t ws_size,
                              hipStream_t stream) {
  (void)in_sizes; (void)n_in; (void)out_size; (void)d_ws; (void)ws_size;
  const float* q = (const float*)d_in[0];
  const float* k = (const float*)d_in[1];
  const float* v = (const float*)d_in[2];
  float* out = (float*)d_out;
  dim3 grid(SEQ / BQ, BHN);   // 16 q-tiles x 16 batch-heads = 256 blocks
  flash_attn_wmma_bf16<<<grid, 512, 0, stream>>>(q, k, v, out);
}
